// GCN_22153441312995
// MI455X (gfx1250) — compile-verified, matrix operand order checked by reading, not run on hardware
//
#include <hip/hip_runtime.h>
#include <stdint.h>
#include <math.h>

// ---------------- problem constants ----------------
#define NFEAT 128   // input feature dim
#define NHID  128   // hidden dim
#define NCLS  40    // classes
#define NCLSP 48    // classes padded to multiple of 16 for WMMA tiles

typedef __attribute__((ext_vector_type(16))) __bf16 v16bf;
typedef __attribute__((ext_vector_type(8)))  __bf16 v8bf;
typedef __attribute__((ext_vector_type(8)))  float  v8f;

// ---------------- small utility kernels ----------------
__global__ void k_fill1(float* __restrict__ p, int n) {
  int i = blockIdx.x * blockDim.x + threadIdx.x;
  if (i < n) p[i] = 1.0f;   // self-loop contributes 1 to every degree
}

__global__ void k_degree(const int* __restrict__ dst, float* __restrict__ deg, int E) {
  int e = blockIdx.x * blockDim.x + threadIdx.x;
  if (e < E) atomicAdd(&deg[dst[e]], 1.0f);
}

__global__ void k_rsqrt(const float* __restrict__ deg, float* __restrict__ dis, int n) {
  int i = blockIdx.x * blockDim.x + threadIdx.x;
  if (i < n) dis[i] = rsqrtf(deg[i]);   // deg >= 1 always (self-loop)
}

__global__ void k_edge_norm(const int* __restrict__ src, const int* __restrict__ dst,
                            const float* __restrict__ dis, float* __restrict__ nrm, int E) {
  int e = blockIdx.x * blockDim.x + threadIdx.x;
  if (e < E) nrm[e] = dis[src[e]] * dis[dst[e]];
}

// fp32 -> packed bf16 (row-major, same layout)
__global__ void k_f32_to_bf16(const float* __restrict__ in, __bf16* __restrict__ out, int n) {
  int i = blockIdx.x * blockDim.x + threadIdx.x;
  if (i < n) out[i] = (__bf16)in[i];
}

// W[K, Nout] fp32  ->  Wt[NoutP, K] bf16 (transposed, rows >= Nout zero-padded)
__global__ void k_w_to_bf16_t(const float* __restrict__ W, __bf16* __restrict__ Wt,
                              int K, int Nout, int NoutP) {
  int i = blockIdx.x * blockDim.x + threadIdx.x;      // over NoutP*K
  if (i < NoutP * K) {
    int nrow = i / K, k = i - nrow * K;
    Wt[i] = (nrow < Nout) ? (__bf16)W[(size_t)k * Nout + nrow] : (__bf16)0.0f;
  }
}

// ---------------- WMMA GEMM: Y[M,COLT*16] = Xb[M,K] @ Wt^T ----------------
// Xb: bf16 row-major [M, K]; Wt: bf16 row-major [COLT*16, K] (pre-transposed W).
// One wave computes one 16x16 fp32 output tile; 8 waves per block; K unrolled.
// Inner loop = 4x global_load_b128 + 1x v_wmma_f32_16x16x32_bf16 per k-step.
template <int K, int COLT>
__global__ void __launch_bounds__(256)
k_gemm_wmma_bf16(const __bf16* __restrict__ Xb, const __bf16* __restrict__ Wt,
                 float* __restrict__ Y, int totalTiles) {
  constexpr int NOUT = COLT * 16;
  int lane = threadIdx.x & 31;
  int wave = threadIdx.x >> 5;
  int t = blockIdx.x * 8 + wave;        // wave-uniform tile id
  if (t >= totalTiles) return;          // whole wave exits together: EXEC all-1s at WMMA
  int rowTile = t / COLT;
  int colTile = t % COLT;
  int m0 = rowTile * 16;
  int n0 = colTile * 16;

  int half = lane >> 4;                 // 0: lanes 0-15, 1: lanes 16-31
  int mr   = lane & 15;

  const v8bf* __restrict__ pa = (const v8bf*)(Xb + (size_t)(m0 + mr) * K + 8 * half);
  const v8bf* __restrict__ pb = (const v8bf*)(Wt + (size_t)(n0 + mr) * K + 8 * half);

  v8f acc = {};                         // fp32 C/D accumulator (8 VGPRs)
#pragma unroll
  for (int kt = 0; kt < K; kt += 32) {
    int c = kt >> 3;                    // chunk index (8 bf16 per chunk)
    // A tile 16x32: lane = row M; elements 0..7 = K kt+8h.., 8..15 = K kt+16+8h..
    v8bf alo = pa[c], ahi = pa[c + 2];
    // B tile 32x16: lane = col N; same K packing (Wt row n is contiguous in K)
    v8bf blo = pb[c], bhi = pb[c + 2];
    v16bf a = __builtin_shufflevector(alo, ahi, 0, 1, 2, 3, 4, 5, 6, 7,
                                      8, 9, 10, 11, 12, 13, 14, 15);
    v16bf b = __builtin_shufflevector(blo, bhi, 0, 1, 2, 3, 4, 5, 6, 7,
                                      8, 9, 10, 11, 12, 13, 14, 15);
    acc = __builtin_amdgcn_wmma_f32_16x16x32_bf16(
        /*neg_a=*/false, a, /*neg_b=*/false, b,
        /*c_mod=*/(short)0, acc, /*reuse_a=*/false, /*reuse_b=*/false);
  }

  // C/D layout: VGPR j holds (M = j + 8*half, N = mr)
#pragma unroll
  for (int j = 0; j < 8; ++j) {
    Y[(size_t)(m0 + j + 8 * half) * NOUT + n0 + mr] = acc[j];
  }
}

// ---------------- aggregation kernels ----------------
// Self-loop term: H = Y * dis[n]^2  (= Y / deg[n])
__global__ void k_selfinit(const float* __restrict__ Y, const float* __restrict__ dis,
                           float* __restrict__ H, int nf, int F) {
  int i = blockIdx.x * blockDim.x + threadIdx.x;
  if (i < nf) {
    int node = i / F;
    float d = dis[node];
    H[i] = Y[i] * d * d;
  }
}

// Edge scatter: H[dst[e], f] += Y[src[e], f] * norm[e]
// Consecutive F threads handle one edge row -> coalesced gather + atomics.
__global__ void k_scatter(const int* __restrict__ src, const int* __restrict__ dst,
                          const float* __restrict__ nrm, const float* __restrict__ Y,
                          float* __restrict__ H, int ef, int F) {
  int i = blockIdx.x * blockDim.x + threadIdx.x;
  if (i < ef) {
    int e = i / F;
    int f = i - e * F;
    float v = Y[(size_t)src[e] * F + f] * nrm[e];
    atomicAdd(&H[(size_t)dst[e] * F + f], v);
  }
}

// bias + ReLU, emitting packed bf16 activations for the next layer's GEMM
__global__ void k_bias_relu_bf16(const float* __restrict__ H, const float* __restrict__ b,
                                 __bf16* __restrict__ Xo, int nf, int F) {
  int i = blockIdx.x * blockDim.x + threadIdx.x;
  if (i < nf) {
    int f = i % F;
    Xo[i] = (__bf16)fmaxf(H[i] + b[f], 0.0f);
  }
}

// relu(H + b3) then log_softmax over the first NCLS columns (H has stride NCLSP)
__global__ void k_logsoftmax(const float* __restrict__ H, const float* __restrict__ b,
                             float* __restrict__ out, int N) {
  int n = blockIdx.x * blockDim.x + threadIdx.x;
  if (n >= N) return;
  float v[NCLS];
  float mx = -1e30f;
#pragma unroll
  for (int c = 0; c < NCLS; ++c) {
    float x = fmaxf(H[(size_t)n * NCLSP + c] + b[c], 0.0f);
    v[c] = x;
    mx = fmaxf(mx, x);
  }
  float s = 0.0f;
#pragma unroll
  for (int c = 0; c < NCLS; ++c) s += expf(v[c] - mx);
  float l = logf(s);
#pragma unroll
  for (int c = 0; c < NCLS; ++c) out[(size_t)n * NCLS + c] = v[c] - mx - l;
}

// ---------------- host-side orchestration ----------------
extern "C" void kernel_launch(void* const* d_in, const int* in_sizes, int n_in,
                              void* d_out, int out_size, void* d_ws, size_t ws_size,
                              hipStream_t stream) {
  const float* X0 = (const float*)d_in[0];   // [N, 128]
  const int*   ei = (const int*)d_in[1];     // [2, E]
  const float* W1 = (const float*)d_in[2];   // [128, 128]
  const float* b1 = (const float*)d_in[3];
  const float* W2 = (const float*)d_in[4];   // [128, 128]
  const float* b2 = (const float*)d_in[5];
  const float* W3 = (const float*)d_in[6];   // [128, 40]
  const float* b3 = (const float*)d_in[7];

  const int N = in_sizes[0] / NFEAT;         // 100000 (multiple of 16)
  const int E = in_sizes[1] / 2;             // 600000
  const int* srcI = ei;
  const int* dstI = ei + E;

  // workspace layout (256B-aligned slices)
  char* p = (char*)d_ws;
  auto alloc = [&](size_t bytes) -> void* {
    void* r = (void*)p;
    p += (bytes + 255) & ~(size_t)255;
    return r;
  };
  float*  deg  = (float*) alloc((size_t)N * 4);
  float*  dis  = (float*) alloc((size_t)N * 4);
  float*  nrm  = (float*) alloc((size_t)E * 4);
  __bf16* Xb   = (__bf16*)alloc((size_t)N * NFEAT * 2);   // bf16 activations
  float*  bufY = (float*) alloc((size_t)N * NFEAT * 4);   // GEMM output
  float*  bufH = (float*) alloc((size_t)N * NFEAT * 4);   // aggregated output
  __bf16* Wt1  = (__bf16*)alloc((size_t)NHID * NFEAT * 2);
  __bf16* Wt2  = (__bf16*)alloc((size_t)NHID * NHID * 2);
  __bf16* Wt3  = (__bf16*)alloc((size_t)NCLSP * NHID * 2);

  const int TB = 256;
  auto blk = [](int n, int tb) { return (n + tb - 1) / tb; };

  // --- normalization coefficients + precision/layout prep ---
  k_fill1     <<<blk(N, TB), TB, 0, stream>>>(deg, N);
  k_degree    <<<blk(E, TB), TB, 0, stream>>>(dstI, deg, E);
  k_rsqrt     <<<blk(N, TB), TB, 0, stream>>>(deg, dis, N);
  k_edge_norm <<<blk(E, TB), TB, 0, stream>>>(srcI, dstI, dis, nrm, E);
  k_f32_to_bf16<<<blk(N * NFEAT, TB), TB, 0, stream>>>(X0, Xb, N * NFEAT);
  k_w_to_bf16_t<<<blk(NHID * NFEAT, TB), TB, 0, stream>>>(W1, Wt1, NFEAT, NHID, NHID);
  k_w_to_bf16_t<<<blk(NHID * NHID, TB), TB, 0, stream>>>(W2, Wt2, NHID, NHID, NHID);
  k_w_to_bf16_t<<<blk(NCLSP * NHID, TB), TB, 0, stream>>>(W3, Wt3, NHID, NCLS, NCLSP);

  const int rowTiles = N / 16;

  // --- layer 1: Xb -> bufY (WMMA) -> bufH (agg) -> Xb (bias+relu, bf16) ---
  {
    int tiles = rowTiles * (NHID / 16);
    k_gemm_wmma_bf16<NFEAT, NHID / 16>
        <<<blk(tiles, 8), 256, 0, stream>>>(Xb, Wt1, bufY, tiles);
    int nf = N * NHID, ef = E * NHID;
    k_selfinit      <<<blk(nf, TB), TB, 0, stream>>>(bufY, dis, bufH, nf, NHID);
    k_scatter       <<<blk(ef, TB), TB, 0, stream>>>(srcI, dstI, nrm, bufY, bufH, ef, NHID);
    k_bias_relu_bf16<<<blk(nf, TB), TB, 0, stream>>>(bufH, b1, Xb, nf, NHID);
  }

  // --- layer 2 ---
  {
    int tiles = rowTiles * (NHID / 16);
    k_gemm_wmma_bf16<NHID, NHID / 16>
        <<<blk(tiles, 8), 256, 0, stream>>>(Xb, Wt2, bufY, tiles);
    int nf = N * NHID, ef = E * NHID;
    k_selfinit      <<<blk(nf, TB), TB, 0, stream>>>(bufY, dis, bufH, nf, NHID);
    k_scatter       <<<blk(ef, TB), TB, 0, stream>>>(srcI, dstI, nrm, bufY, bufH, ef, NHID);
    k_bias_relu_bf16<<<blk(nf, TB), TB, 0, stream>>>(bufH, b2, Xb, nf, NHID);
  }

  // --- layer 3 (48-wide padded) + log_softmax ---
  {
    int tiles = rowTiles * (NCLSP / 16);
    k_gemm_wmma_bf16<NHID, NCLSP / 16>
        <<<blk(tiles, 8), 256, 0, stream>>>(Xb, Wt3, bufY, tiles);
    int nf = N * NCLSP, ef = E * NCLSP;
    k_selfinit  <<<blk(nf, TB), TB, 0, stream>>>(bufY, dis, bufH, nf, NCLSP);
    k_scatter   <<<blk(ef, TB), TB, 0, stream>>>(srcI, dstI, nrm, bufY, bufH, ef, NCLSP);
    k_logsoftmax<<<blk(N, TB), TB, 0, stream>>>(bufH, b3, (float*)d_out, N);
  }
}
